// TimeSeriesVAE_82162724372614
// MI455X (gfx1250) — compile-verified, matrix operand order checked by reading, not run on hardware
//
#include <hip/hip_runtime.h>
#include <hip/hip_bf16.h>

// ---- problem constants ----
#define H_   512
#define B_   128
#define T_   256
#define D_   8
#define L_   64
#define FOURH 2048
#define NBLK 32          // persistent grid: 32 blocks x 256 threads = 256 waves
#define HB   (B_ * H_)   // elements in one (B,H) buffer = 65536

// LDS weight slice: 64 rows (4 gates x 16 cols) x 512 K, row-padded to 520
// elems (1040B = 260 dwords == 4 mod 64 banks -> conflict-free b128 reads).
#define WROW 520
#define WSLOT (64 * WROW)            // 33280 elems = 66560 B per matrix slice
#define PROW 40                      // padded row for the 32-wide Wih0 slice
#define PADOFF (4 * WSLOT)           // pad slice after 4 weight slots
#define LDS_ELEMS (PADOFF + 64 * PROW)   // 135680 elems = 271360 bytes

typedef __attribute__((ext_vector_type(8)))  float   v8f;
typedef __attribute__((ext_vector_type(16))) __bf16  v16bf;

struct alignas(16) U4 { unsigned int a, b, c, d; };
union FragU { U4 u[2]; v16bf v; };

#define WMMA_BF16(a, b, c) __builtin_amdgcn_wmma_f32_16x16x32_bf16( \
    false, (a), false, (b), (short)0, (c), false, false)

// ---------------- small device helpers ----------------
static __device__ __forceinline__ unsigned short f2bf_d(float f) {
  unsigned u = __float_as_uint(f);
  u += 0x7FFFu + ((u >> 16) & 1u);          // round-to-nearest-even
  return (unsigned short)(u >> 16);
}
static __device__ __forceinline__ float bf2f_d(unsigned short s) {
  return __uint_as_float(((unsigned)s) << 16);
}
static __device__ __forceinline__ float sig_f(float x) {
  return 1.0f / (1.0f + __expf(-x));
}
static __device__ __forceinline__ float tanh_f(float x) {
  float e = __expf(2.0f * x);
  return 1.0f - 2.0f / (e + 1.0f);          // safe at +-inf
}
static __device__ __forceinline__ v8f zero8() {
  v8f z;
#pragma unroll
  for (int e = 0; e < 8; ++e) z[e] = 0.0f;
  return z;
}

// Load a 16x32 bf16 WMMA fragment from a row-major source (global or LDS).
// ISA layout: lanes 0-15 -> rows 0-15, K chunks [0..8)+[16..24);
//             lanes 16-31 -> rows 0-15, K chunks [8..16)+[24..32).
static __device__ __forceinline__ v16bf loadFrag(const unsigned short* base,
                                                 int rowStride, int r, int hi) {
  const unsigned short* p = base + r * rowStride + hi * 8;
  FragU f;
  f.u[0] = *reinterpret_cast<const U4*>(p);
  f.u[1] = *reinterpret_cast<const U4*>(p + 16);
  return f.v;
}

// Fused dual-stream GEMM: gates += A0 x W0 + A1 x W1 over K=512, both into the
// same 4 gate accumulators. Manual ping-pong software pipeline: two disjoint
// fragment sets alternate (each written once / read once per iteration -> no
// register-rotation copies), with the next tile's 2 global + 8 LDS loads in
// flight while the current tile's 8 WMMAs issue.
static __device__ __forceinline__ void gemm2K512_lds(v8f acc[4],
                                                     const unsigned short* a0Base,
                                                     const unsigned short* a1Base,
                                                     int aStride,
                                                     const unsigned short* w0,
                                                     const unsigned short* w1,
                                                     int r, int hi) {
  v16bf S0[10], S1[10];
  auto loadSet = [&](v16bf* S, int kk) {
    S[0] = loadFrag(a0Base + kk, aStride, r, hi);
    S[1] = loadFrag(a1Base + kk, aStride, r, hi);
    S[2] = loadFrag(w0 + 0 * 16 * WROW + kk, WROW, r, hi);
    S[3] = loadFrag(w0 + 1 * 16 * WROW + kk, WROW, r, hi);
    S[4] = loadFrag(w0 + 2 * 16 * WROW + kk, WROW, r, hi);
    S[5] = loadFrag(w0 + 3 * 16 * WROW + kk, WROW, r, hi);
    S[6] = loadFrag(w1 + 0 * 16 * WROW + kk, WROW, r, hi);
    S[7] = loadFrag(w1 + 1 * 16 * WROW + kk, WROW, r, hi);
    S[8] = loadFrag(w1 + 2 * 16 * WROW + kk, WROW, r, hi);
    S[9] = loadFrag(w1 + 3 * 16 * WROW + kk, WROW, r, hi);
  };
  auto compute = [&](const v16bf* S) {
    acc[0] = WMMA_BF16(S[0], S[2], acc[0]);
    acc[1] = WMMA_BF16(S[0], S[3], acc[1]);
    acc[2] = WMMA_BF16(S[0], S[4], acc[2]);
    acc[3] = WMMA_BF16(S[0], S[5], acc[3]);
    acc[0] = WMMA_BF16(S[1], S[6], acc[0]);
    acc[1] = WMMA_BF16(S[1], S[7], acc[1]);
    acc[2] = WMMA_BF16(S[1], S[8], acc[2]);
    acc[3] = WMMA_BF16(S[1], S[9], acc[3]);
  };
  loadSet(S0, 0);                       // tile 0
#pragma unroll 1
  for (int it = 0; it < 7; ++it) {      // tiles 2it .. 2it+1  (0..13)
    int k = it * 64;
    loadSet(S1, k + 32);
    compute(S0);
    loadSet(S0, k + 64);
    compute(S1);
  }
  loadSet(S1, H_ - 32);                 // tile 15
  compute(S0);                          // tile 14
  compute(S1);                          // tile 15
}

// Single-stream K=512 GEMM (encoder layer0 recurrent part), same ping-pong.
static __device__ __forceinline__ void gemmK512_lds(v8f acc[4],
                                                    const unsigned short* aBase,
                                                    int aStride,
                                                    const unsigned short* wl,
                                                    int r, int hi) {
  v16bf S0[5], S1[5];
  auto loadSet = [&](v16bf* S, int kk) {
    S[0] = loadFrag(aBase + kk, aStride, r, hi);
    S[1] = loadFrag(wl + 0 * 16 * WROW + kk, WROW, r, hi);
    S[2] = loadFrag(wl + 1 * 16 * WROW + kk, WROW, r, hi);
    S[3] = loadFrag(wl + 2 * 16 * WROW + kk, WROW, r, hi);
    S[4] = loadFrag(wl + 3 * 16 * WROW + kk, WROW, r, hi);
  };
  auto compute = [&](const v16bf* S) {
    acc[0] = WMMA_BF16(S[0], S[1], acc[0]);
    acc[1] = WMMA_BF16(S[0], S[2], acc[1]);
    acc[2] = WMMA_BF16(S[0], S[3], acc[2]);
    acc[3] = WMMA_BF16(S[0], S[4], acc[3]);
  };
  loadSet(S0, 0);
#pragma unroll 1
  for (int it = 0; it < 7; ++it) {
    int k = it * 64;
    loadSet(S1, k + 32);
    compute(S0);
    loadSet(S0, k + 64);
    compute(S1);
  }
  loadSet(S1, H_ - 32);
  compute(S0);
  compute(S1);
}

static __device__ __forceinline__ void initBias(v8f acc[4], const float* b, int laneN) {
#pragma unroll
  for (int g = 0; g < 4; ++g) {
    float bv = b[g * H_ + laneN];
#pragma unroll
    for (int e = 0; e < 8; ++e) acc[g][e] = bv;
  }
}

// LSTM cell on the 4 gate accumulators; c kept in registers; h stored bf16.
// C/D layout: element e of lane -> (M = m0 + e + hi*8, N = laneN).
static __device__ __forceinline__ void cellStore(v8f acc[4], v8f& c,
                                                 unsigned short* hdst0,
                                                 unsigned short* hdst1,
                                                 int m0, int laneN, int hi) {
#pragma unroll
  for (int e = 0; e < 8; ++e) {
    float ig = sig_f(acc[0][e]);
    float fg = sig_f(acc[1][e]);
    float gg = tanh_f(acc[2][e]);
    float og = sig_f(acc[3][e]);
    float cn = fg * c[e] + ig * gg;
    c[e] = cn;
    unsigned short hb = f2bf_d(og * tanh_f(cn));
    int row = m0 + e + (hi ? 8 : 0);
    hdst0[row * H_ + laneN] = hb;
    if (hdst1) hdst1[row * H_ + laneN] = hb;
  }
}

// Stage this block's slice of a (2048 x 512) bf16 weight into an LDS slot:
// rows g*512+n0+rr (g<4, rr<16) -> dst[(g*16+rr)*WROW + k].
static __device__ __forceinline__ void stageWeight(unsigned short* dst,
                                                   const unsigned short* w, int n0) {
  for (int idx = threadIdx.x; idx < 64 * 64; idx += 256) {
    int row = idx >> 6;            // 0..63
    int kc  = (idx & 63) * 8;      // 0..504
    int g = row >> 4, rr = row & 15;
    U4 s = *reinterpret_cast<const U4*>(w + (g * H_ + n0 + rr) * H_ + kc);
    *reinterpret_cast<U4*>(dst + row * WROW + kc) = s;
  }
}
// Stage the (2048 x 32) padded Wih0 slice into LDS (rows padded to PROW).
static __device__ __forceinline__ void stagePad(unsigned short* dst,
                                                const unsigned short* w, int n0) {
  for (int idx = threadIdx.x; idx < 64 * 4; idx += 256) {
    int row = idx >> 2;
    int kc  = (idx & 3) * 8;
    int g = row >> 4, rr = row & 15;
    U4 s = *reinterpret_cast<const U4*>(w + ((g * H_ + n0 + rr) * 32) + kc);
    *reinterpret_cast<U4*>(dst + row * PROW + kc) = s;
  }
}

// sense-reversing device-wide barrier (all NBLK blocks co-resident)
static __device__ __forceinline__ void gridBarrier(unsigned* cnt, unsigned* gen) {
  __syncthreads();
  if (threadIdx.x == 0) {
    __threadfence();
    unsigned g = __hip_atomic_load(gen, __ATOMIC_RELAXED, __HIP_MEMORY_SCOPE_AGENT);
    unsigned arr = __hip_atomic_fetch_add(cnt, 1u, __ATOMIC_ACQ_REL, __HIP_MEMORY_SCOPE_AGENT);
    if (arr == NBLK - 1) {
      __hip_atomic_store(cnt, 0u, __ATOMIC_RELAXED, __HIP_MEMORY_SCOPE_AGENT);
      __hip_atomic_store(gen, g + 1u, __ATOMIC_RELEASE, __HIP_MEMORY_SCOPE_AGENT);
    } else {
      while (__hip_atomic_load(gen, __ATOMIC_ACQUIRE, __HIP_MEMORY_SCOPE_AGENT) == g)
        __builtin_amdgcn_s_sleep(2);
    }
  }
  __syncthreads();
}

// ---------------- kernel arg structs ----------------
struct PrepArgs {
  const float* wsrc[7];           // encWhh0, encWih1, encWhh1, decWih0, decWhh0, decWih1, decWhh1
  unsigned short* wdst[7];
  const float* wih0;  unsigned short* wih0pad;   // (2048,8) -> (2048,32)
  const float* x;     unsigned short* xpad;      // (B,T,8)  -> (B,T,32)
  unsigned short* zA; // e0h buf0
  unsigned short* zB; // e1h buf0
  unsigned short* zC; // zbuf
  unsigned* bar;      // 2 counters
};

struct VaeArgs {
  const float *eps;
  const float *enc_b0, *enc_b1;
  const float *W_mu, *b_mu, *W_lv, *b_lv, *W_di, *b_di;
  const float *dec_b0, *dec_b1, *W_out, *b_out;
  const unsigned short *wEncWhh0, *wEncWih1, *wEncWhh1;
  const unsigned short *wDecWih0, *wDecWhh0, *wDecWih1, *wDecWhh1;
  const unsigned short *wih0pad, *xpad;
  unsigned short *e0h, *e1h, *y0b, *d0h, *d1h, *zbuf;  // [2]x(B,H) each (zbuf single)
  float *zlat;
  unsigned *bar;
  float *out;
};

// ---------------- prep: f32->bf16 weights, pad x, zero state ----------------
__global__ void vae_prep(PrepArgs P) {
  const long NW = 7L * FOURH * H_;     // 7,340,032
  const long NP = (long)FOURH * 32;    // 65,536
  const long NX = (long)B_ * T_ * 32;  // 1,048,576
  const long NZ = 3L * HB;
  const long total = NW + NP + NX + NZ + 2;
  long stride = (long)gridDim.x * blockDim.x;
  for (long i = (long)blockIdx.x * blockDim.x + threadIdx.x; i < total; i += stride) {
    if (i < NW) {
      long m = i / ((long)FOURH * H_), e = i % ((long)FOURH * H_);
      P.wdst[m][e] = f2bf_d(P.wsrc[m][e]);
    } else if (i < NW + NP) {
      long j = i - NW; long n = j >> 5, col = j & 31;
      P.wih0pad[j] = (col < D_) ? f2bf_d(P.wih0[n * D_ + col]) : (unsigned short)0;
    } else if (i < NW + NP + NX) {
      long j = i - NW - NP; long row = j >> 5, col = j & 31;
      P.xpad[j] = (col < D_) ? f2bf_d(P.x[row * D_ + col]) : (unsigned short)0;
    } else if (i < NW + NP + NX + NZ) {
      long j = i - NW - NP - NX;
      if (j < HB)            P.zA[j] = 0;
      else if (j < 2 * HB)   P.zB[j - HB] = 0;
      else                   P.zC[j - 2 * HB] = 0;
    } else {
      P.bar[i - (NW + NP + NX + NZ)] = 0u;
    }
  }
}

// ---------------- output projection: y[b,t,:] = h1 @ W_out.T + b_out ----------------
static __device__ __forceinline__ void outProj(const VaeArgs& A, int flat, int tOut,
                                               const unsigned short* h1) {
  int b = flat >> 3, d = flat & 7;
  float s = A.b_out[d];
  const unsigned short* hp = h1 + b * H_;
  const float* wp = A.W_out + d * H_;
  for (int k = 0; k < H_; ++k) s += bf2f_d(hp[k]) * wp[k];
  A.out[(b * T_ + tOut) * D_ + d] = s;
}

// ---------------- persistent VAE kernel ----------------
__global__ __launch_bounds__(256) void vae_persistent(VaeArgs A) {
  extern __shared__ __align__(16) unsigned short smem[];
  const int lane  = threadIdx.x & 31;
  const int r     = lane & 15;
  const int hi    = lane >> 4;
  const int warp  = threadIdx.x >> 5;       // 0..7
  const int m0    = warp * 16;              // batch tile: per-wave within block
  const int n0    = blockIdx.x * 16;        // hidden tile: per-block (8 waves share)
  const int laneN = n0 + r;
  const int flat  = blockIdx.x * 256 + threadIdx.x;
  unsigned* cnt = A.bar;
  unsigned* gen = A.bar + 1;

  unsigned short* ldsW0  = smem;                 // slot 0
  unsigned short* ldsW1  = smem + WSLOT;         // slot 1
  unsigned short* ldsW2  = smem + 2 * WSLOT;     // slot 2
  unsigned short* ldsW3  = smem + 3 * WSLOT;     // slot 3
  unsigned short* ldsPad = smem + PADOFF;        // Wih0 padded slice

  // ---- stage encoder weights into LDS (once, reused for all 256 steps) ----
  stageWeight(ldsW0, A.wEncWhh0, n0);
  stageWeight(ldsW1, A.wEncWih1, n0);
  stageWeight(ldsW2, A.wEncWhh1, n0);
  stagePad(ldsPad, A.wih0pad, n0);
  __syncthreads();

  // ---- encoder: layer0 step p pipelined with layer1 step p-1 (1 barrier/step) ----
  v8f e0c = zero8(), e1c = zero8();
  for (int p = 0; p <= T_; ++p) {
    if (p < T_) {                                   // enc layer0, step t=p
      int t = p;
      v8f acc[4];
      initBias(acc, A.enc_b0, laneN);
      {   // input projection: one K=32 tile from padded x, weights in LDS
        v16bf a = loadFrag(A.xpad + (m0 * T_ + t) * 32, T_ * 32, r, hi);
#pragma unroll
        for (int g = 0; g < 4; ++g) {
          v16bf b = loadFrag(ldsPad + (g * 16) * PROW, PROW, r, hi);
          acc[g] = WMMA_BF16(a, b, acc[g]);
        }
      }
      gemmK512_lds(acc, A.e0h + (t & 1) * HB + m0 * H_, H_, ldsW0, r, hi);
      cellStore(acc, e0c, A.e0h + ((t + 1) & 1) * HB, A.y0b + (t & 1) * HB, m0, laneN, hi);
    }
    if (p >= 1) {                                   // enc layer1, step t=p-1
      int t = p - 1;
      v8f acc[4];
      initBias(acc, A.enc_b1, laneN);
      gemm2K512_lds(acc,
                    A.y0b + (t & 1) * HB + m0 * H_,   // input stream (ys0[t])
                    A.e1h + (t & 1) * HB + m0 * H_,   // recurrent stream
                    H_, ldsW1, ldsW2, r, hi);
      cellStore(acc, e1c, A.e1h + ((t + 1) & 1) * HB, nullptr, m0, laneN, hi);
    }
    gridBarrier(cnt, gen);
  }

  // ---- latent: mu/logvar/z + decoder init (block 0 only); h_last = e1h buf0 ----
  if (blockIdx.x == 0) {
    const unsigned short* hl = A.e1h;   // buffer ((T)&1)==0
    for (int idx = threadIdx.x; idx < B_ * L_; idx += 256) {
      int b = idx >> 6, l = idx & 63;
      float mu = A.b_mu[l], lv = A.b_lv[l];
      const unsigned short* hp = hl + b * H_;
      const float* wm = A.W_mu + l * H_;
      const float* wl = A.W_lv + l * H_;
      for (int k = 0; k < H_; ++k) {
        float hv = bf2f_d(hp[k]);
        mu += hv * wm[k];
        lv += hv * wl[k];
      }
      A.out[B_ * T_ * D_ + idx]           = mu;   // mu block
      A.out[B_ * T_ * D_ + B_ * L_ + idx] = lv;   // logvar block
      A.zlat[idx] = mu + A.eps[idx] * __expf(0.5f * lv);
    }
    __syncthreads();
    for (int idx = threadIdx.x; idx < B_ * H_; idx += 256) {
      int b = idx >> 9, hc = idx & (H_ - 1);
      float acc = A.b_di[hc];
      const float* zp = A.zlat + b * L_;
      const float* wp = A.W_di + hc * L_;
      for (int k = 0; k < L_; ++k) acc += zp[k] * wp[k];
      unsigned short hb = f2bf_d(acc);
      A.d0h[idx] = hb;   // both decoder layers init h = h0
      A.d1h[idx] = hb;
    }
  }
  gridBarrier(cnt, gen);

  // ---- stage decoder weights into LDS (reuse encoder slots) ----
  stageWeight(ldsW0, A.wDecWih0, n0);
  stageWeight(ldsW1, A.wDecWhh0, n0);
  stageWeight(ldsW2, A.wDecWih1, n0);
  stageWeight(ldsW3, A.wDecWhh1, n0);
  __syncthreads();

  // ---- decoder: coupled 2-layer autoregressive loop (2 barriers/step) ----
  v8f d0c = zero8(), d1c = zero8();
  for (int t = 0; t < T_; ++t) {
    // output projection for step t-1 (h1_{t-1} in buf t&1), overlapped with GEMM0
    if (t >= 1 && flat < B_ * D_) outProj(A, flat, t - 1, A.d1h + (t & 1) * HB);
    {   // layer 0: input = h1_{t-1} (zeros at t=0), recurrent = h0_{t-1}
      v8f acc[4];
      initBias(acc, A.dec_b0, laneN);
      const unsigned short* inp = (t == 0) ? A.zbuf : (A.d1h + (t & 1) * HB);
      gemm2K512_lds(acc,
                    inp + m0 * H_,
                    A.d0h + (t & 1) * HB + m0 * H_,
                    H_, ldsW0, ldsW1, r, hi);
      cellStore(acc, d0c, A.d0h + ((t + 1) & 1) * HB, nullptr, m0, laneN, hi);
    }
    gridBarrier(cnt, gen);
    {   // layer 1: input = h0_t (just published), recurrent = h1_{t-1}
      v8f acc[4];
      initBias(acc, A.dec_b1, laneN);
      gemm2K512_lds(acc,
                    A.d0h + ((t + 1) & 1) * HB + m0 * H_,
                    A.d1h + (t & 1) * HB + m0 * H_,
                    H_, ldsW2, ldsW3, r, hi);
      cellStore(acc, d1c, A.d1h + ((t + 1) & 1) * HB, nullptr, m0, laneN, hi);
    }
    gridBarrier(cnt, gen);
  }
  if (flat < B_ * D_) outProj(A, flat, T_ - 1, A.d1h);  // h1_{255} in buf 0
}

// ---------------- host launcher ----------------
extern "C" void kernel_launch(void* const* d_in, const int* in_sizes, int n_in,
                              void* d_out, int out_size, void* d_ws, size_t ws_size,
                              hipStream_t stream) {
  (void)in_sizes; (void)n_in; (void)out_size; (void)ws_size;
  const float* x        = (const float*)d_in[0];
  const float* eps      = (const float*)d_in[1];
  const float* enc_Wih0 = (const float*)d_in[2];
  const float* enc_Whh0 = (const float*)d_in[3];
  const float* enc_b0   = (const float*)d_in[4];
  const float* enc_Wih1 = (const float*)d_in[5];
  const float* enc_Whh1 = (const float*)d_in[6];
  const float* enc_b1   = (const float*)d_in[7];
  const float* W_mu     = (const float*)d_in[8];
  const float* b_mu     = (const float*)d_in[9];
  const float* W_lv     = (const float*)d_in[10];
  const float* b_lv     = (const float*)d_in[11];
  const float* W_di     = (const float*)d_in[12];
  const float* b_di     = (const float*)d_in[13];
  const float* dec_Wih0 = (const float*)d_in[14];
  const float* dec_Whh0 = (const float*)d_in[15];
  const float* dec_b0   = (const float*)d_in[16];
  const float* dec_Wih1 = (const float*)d_in[17];
  const float* dec_Whh1 = (const float*)d_in[18];
  const float* dec_b1   = (const float*)d_in[19];
  const float* W_out    = (const float*)d_in[20];
  const float* b_out    = (const float*)d_in[21];

  char* ws = (char*)d_ws;
  size_t off = 0;
  auto take = [&](size_t bytes) -> char* {
    char* p = ws + off;
    off += (bytes + 255) & ~(size_t)255;
    return p;
  };
  const size_t WBYTES = (size_t)FOURH * H_ * 2;     // 2 MB bf16 weight
  unsigned short* wEncWhh0 = (unsigned short*)take(WBYTES);
  unsigned short* wEncWih1 = (unsigned short*)take(WBYTES);
  unsigned short* wEncWhh1 = (unsigned short*)take(WBYTES);
  unsigned short* wDecWih0 = (unsigned short*)take(WBYTES);
  unsigned short* wDecWhh0 = (unsigned short*)take(WBYTES);
  unsigned short* wDecWih1 = (unsigned short*)take(WBYTES);
  unsigned short* wDecWhh1 = (unsigned short*)take(WBYTES);
  unsigned short* wih0pad  = (unsigned short*)take((size_t)FOURH * 32 * 2);
  unsigned short* xpad     = (unsigned short*)take((size_t)B_ * T_ * 32 * 2);
  unsigned short* e0h      = (unsigned short*)take(2 * HB * 2);
  unsigned short* e1h      = (unsigned short*)take(2 * HB * 2);
  unsigned short* y0b      = (unsigned short*)take(2 * HB * 2);
  unsigned short* d0h      = (unsigned short*)take(2 * HB * 2);
  unsigned short* d1h      = (unsigned short*)take(2 * HB * 2);
  unsigned short* zbuf     = (unsigned short*)take(HB * 2);
  float*          zlat     = (float*)take((size_t)B_ * L_ * 4);
  unsigned*       bar      = (unsigned*)take(256);

  PrepArgs P;
  P.wsrc[0] = enc_Whh0; P.wdst[0] = wEncWhh0;
  P.wsrc[1] = enc_Wih1; P.wdst[1] = wEncWih1;
  P.wsrc[2] = enc_Whh1; P.wdst[2] = wEncWhh1;
  P.wsrc[3] = dec_Wih0; P.wdst[3] = wDecWih0;
  P.wsrc[4] = dec_Whh0; P.wdst[4] = wDecWhh0;
  P.wsrc[5] = dec_Wih1; P.wdst[5] = wDecWih1;
  P.wsrc[6] = dec_Whh1; P.wdst[6] = wDecWhh1;
  P.wih0 = enc_Wih0; P.wih0pad = wih0pad;
  P.x = x; P.xpad = xpad;
  P.zA = e0h; P.zB = e1h; P.zC = zbuf;
  P.bar = bar;
  vae_prep<<<4096, 256, 0, stream>>>(P);

  VaeArgs A;
  A.eps = eps;
  A.enc_b0 = enc_b0; A.enc_b1 = enc_b1;
  A.W_mu = W_mu; A.b_mu = b_mu; A.W_lv = W_lv; A.b_lv = b_lv;
  A.W_di = W_di; A.b_di = b_di;
  A.dec_b0 = dec_b0; A.dec_b1 = dec_b1;
  A.W_out = W_out; A.b_out = b_out;
  A.wEncWhh0 = wEncWhh0; A.wEncWih1 = wEncWih1; A.wEncWhh1 = wEncWhh1;
  A.wDecWih0 = wDecWih0; A.wDecWhh0 = wDecWhh0; A.wDecWih1 = wDecWih1; A.wDecWhh1 = wDecWhh1;
  A.wih0pad = wih0pad; A.xpad = xpad;
  A.e0h = e0h; A.e1h = e1h; A.y0b = y0b; A.d0h = d0h; A.d1h = d1h; A.zbuf = zbuf;
  A.zlat = zlat; A.bar = bar;
  A.out = (float*)d_out;
  vae_persistent<<<NBLK, 256, LDS_ELEMS * 2, stream>>>(A);
}
